// GCN_33208687133423
// MI455X (gfx1250) — matrix operand order repaired and checked by reading, not compiled, not run on hardware
//
#include <hip/hip_runtime.h>

// ---------------- problem constants (from reference) ----------------
#define NNODES 100000
#define NEDGES 1600000
#define FEAT   128
#define OUTF   64
#define NCLUST 60000

// ---------------- workspace layout (bytes) ----------------
// h1  : NNODES*FEAT*4      = 51,200,000
// h2  : NNODES*FEAT*4      = 51,200,000
// A   : NCLUST*FEAT*2      = 15,360,000   (bf16 activations for WMMA)
// HC  : NCLUST*FEAT*4      = 30,720,000   (f32 GEMM output; GEMM2 uses subset)
// W1T : FEAT*FEAT*2        = 32,768       (bf16, transposed [n][k])
// W2T : FEAT*OUTF*2        = 16,384
#define OFF_H1   ((size_t)0)
#define OFF_H2   ((size_t)51200000)
#define OFF_A    ((size_t)102400000)
#define OFF_HC   ((size_t)117760000)
#define OFF_W1T  ((size_t)148480000)
#define OFF_W2T  ((size_t)148513024)

typedef __attribute__((ext_vector_type(16))) __bf16 v16bf;
typedef __attribute__((ext_vector_type(8)))  float  v8f;

static __device__ __forceinline__ unsigned short f32_to_bf16(float f) {
    unsigned int u = __float_as_uint(f);
    unsigned int r = u + 0x7FFFu + ((u >> 16) & 1u);   // round-to-nearest-even
    return (unsigned short)(r >> 16);
}
static __device__ __forceinline__ unsigned int pack_bf16x2(float lo, float hi) {
    return (unsigned int)f32_to_bf16(lo) | ((unsigned int)f32_to_bf16(hi) << 16);
}

// ---------------- zero fill ----------------
__global__ void k_zero_f32(float* __restrict__ p, long long n) {
    long long i = (long long)blockIdx.x * blockDim.x + threadIdx.x;
    if (i < n) p[i] = 0.0f;
}

// ---------------- weighted scatter-add propagate ----------------
// one wave per edge; each lane handles 4 consecutive features (float4 load,
// 4x global_atomic_add_f32 scatter). x and h are L2-resident (51MB << 192MB).
__global__ void k_propagate(const float* __restrict__ x,
                            const int* __restrict__ src,
                            const int* __restrict__ dst,
                            const float* __restrict__ wgt,
                            float* __restrict__ h) {
    long long gid = (long long)blockIdx.x * blockDim.x + threadIdx.x;
    int e = (int)(gid >> 5);
    if (e >= NEDGES) return;
    int lane = (int)(gid & 31);
    int s = src[e], d = dst[e];
    float w = wgt[e];
    const float4 v = *(const float4*)(x + (long long)s * FEAT + lane * 4);
    float* hp = h + (long long)d * FEAT + lane * 4;
    atomicAdd(hp + 0, v.x * w);
    atomicAdd(hp + 1, v.y * w);
    atomicAdd(hp + 2, v.z * w);
    atomicAdd(hp + 3, v.w * w);
}

// ---------------- gather rows + cast f32 -> bf16 ----------------
// thread handles 8 elements: two float4 reads, one 16B (uint4) bf16 write.
__global__ void k_gather_cast_bf16(unsigned short* __restrict__ dst,
                                   const float* __restrict__ srcmat,
                                   const int* __restrict__ idx) {
    long long gid = (long long)blockIdx.x * blockDim.x + threadIdx.x;
    if (gid >= (long long)NCLUST * (FEAT / 8)) return;
    int c = (int)(gid >> 4);
    int k = (int)(gid & 15) * 8;
    const float* r = srcmat + (long long)idx[c] * FEAT + k;
    float4 a = *(const float4*)(r);
    float4 b = *(const float4*)(r + 4);
    uint4 o;
    o.x = pack_bf16x2(a.x, a.y);
    o.y = pack_bf16x2(a.z, a.w);
    o.z = pack_bf16x2(b.x, b.y);
    o.w = pack_bf16x2(b.z, b.w);
    *(uint4*)(dst + (long long)c * FEAT + k) = o;
}

// ---------------- transpose-cast weights: W[k][n] f32 -> WT[n][k] bf16 ------
__global__ void k_transpose_cast(unsigned short* __restrict__ WT,
                                 const float* __restrict__ W,
                                 int ncols /*N*/, int K) {
    int gid = blockIdx.x * blockDim.x + threadIdx.x;
    if (gid >= ncols * K) return;
    int n = gid / K, k = gid % K;
    WT[gid] = f32_to_bf16(W[k * ncols + n]);
}

// ---------------- bf16 WMMA GEMM: Out[C,N] = act(A[C,128] @ W + bias) -------
// One wave per 16x16 tile, K=128 = 4 x v_wmma_f32_16x16x32_bf16.
// A fragment (16x32, 16-bit): lane L<16 -> row L, K={k0..k0+7, k0+16..k0+23};
// lane L>=16 -> row L-16, K shifted by 8.  (ISA 7.12.2 layout)
// B fragment (32x16): same K order as A (lane L<16: K={k0..k0+7, k0+16..k0+23},
// lane L>=16: shifted by 8) from WT[n][k], so all loads are 16B global_load_b128.
template <int NCOLS, bool RELU>
__global__ void k_gemm_wmma(const unsigned short* __restrict__ A,
                            const unsigned short* __restrict__ WT,
                            const float* __restrict__ bias,
                            float* __restrict__ Out,
                            int Mrows) {
    const int K = FEAT;
    int wave = (int)(((long long)blockIdx.x * blockDim.x + threadIdx.x) >> 5);
    int lane = threadIdx.x & 31;
    const int tilesN = NCOLS / 16;
    int tileM = wave / tilesN;
    int tileN = wave % tilesN;
    if (tileM * 16 >= Mrows) return;

    int half = lane >> 4;     // 0: lanes 0-15, 1: lanes 16-31
    int l15  = lane & 15;

    const unsigned short* Arow = A  + (long long)(tileM * 16 + l15) * K;
    const unsigned short* Bcol = WT + (long long)(tileN * 16 + l15) * K;

    v8f acc = {};
#pragma unroll
    for (int kb = 0; kb < 4; ++kb) {
        union { v16bf v; uint4 q[2]; } af, bf;
        int ka = kb * 32 + half * 8;
        af.q[0] = *(const uint4*)(Arow + ka);
        af.q[1] = *(const uint4*)(Arow + ka + 16);
        int kbb = kb * 32 + half * 8;
        bf.q[0] = *(const uint4*)(Bcol + kbb);
        bf.q[1] = *(const uint4*)(Bcol + kbb + 16);
        acc = __builtin_amdgcn_wmma_f32_16x16x32_bf16(
            /*neg_a=*/false, af.v, /*neg_b=*/false, bf.v,
            /*c_mod=*/(short)0, acc, /*reuse_a=*/false, /*reuse_b=*/false);
    }

    float bv = bias[tileN * 16 + l15];
    // C/D layout: VGPR r -> row tileM*16 + half*8 + r, col tileN*16 + l15
    float* outp = Out + (long long)(tileM * 16 + half * 8) * NCOLS
                      + tileN * 16 + l15;
#pragma unroll
    for (int r = 0; r < 8; ++r) {
        float v = acc[r] + bv;
        if (RELU) v = fmaxf(v, 0.0f);
        outp[(long long)r * NCOLS] = v;
    }
}

// ---------------- gather f32 rows (128 cols), float4 per thread -------------
__global__ void k_gather_f32_128(float* __restrict__ dst,
                                 const float* __restrict__ srcmat,
                                 const int* __restrict__ idx) {
    long long gid = (long long)blockIdx.x * blockDim.x + threadIdx.x;
    if (gid >= (long long)NNODES * (FEAT / 4)) return;
    int n = (int)(gid >> 5);
    int k = (int)(gid & 31) * 4;
    *(float4*)(dst + (long long)n * FEAT + k) =
        *(const float4*)(srcmat + (long long)idx[n] * FEAT + k);
}

// ---------------- in-place log_softmax over rows of [C, 64] -----------------
// one wave32 per row, 2 elements per lane, shfl_xor tree reductions.
__global__ void k_logsoftmax64(float* __restrict__ M, int rows) {
    long long gid = (long long)blockIdx.x * blockDim.x + threadIdx.x;
    int row = (int)(gid >> 5);
    if (row >= rows) return;
    int lane = (int)(gid & 31);
    float* p = M + (long long)row * OUTF;
    float a = p[lane];
    float b = p[lane + 32];
    float m = fmaxf(a, b);
#pragma unroll
    for (int i = 16; i > 0; i >>= 1) m = fmaxf(m, __shfl_xor(m, i, 32));
    float s = __expf(a - m) + __expf(b - m);
#pragma unroll
    for (int i = 16; i > 0; i >>= 1) s += __shfl_xor(s, i, 32);
    float ls = m + __logf(s);
    p[lane]      = a - ls;
    p[lane + 32] = b - ls;
}

// ---------------- gather f32 rows (64 cols) to output -----------------------
__global__ void k_gather_f32_64(float* __restrict__ dst,
                                const float* __restrict__ srcmat,
                                const int* __restrict__ idx) {
    long long gid = (long long)blockIdx.x * blockDim.x + threadIdx.x;
    if (gid >= (long long)NNODES * (OUTF / 4)) return;
    int n = (int)(gid >> 4);
    int k = (int)(gid & 15) * 4;
    *(float4*)(dst + (long long)n * OUTF + k) =
        *(const float4*)(srcmat + (long long)idx[n] * OUTF + k);
}

// ---------------------------------------------------------------------------
extern "C" void kernel_launch(void* const* d_in, const int* in_sizes, int n_in,
                              void* d_out, int out_size, void* d_ws, size_t ws_size,
                              hipStream_t stream) {
    const float* x    = (const float*)d_in[0];
    const int*   ei   = (const int*)  d_in[1];   // [2,E] flat: src=ei, dst=ei+E
    const float* ew   = (const float*)d_in[2];
    const int*   rep1 = (const int*)  d_in[3];
    const int*   idx1 = (const int*)  d_in[4];
    const int*   rep2 = (const int*)  d_in[5];
    const int*   idx2 = (const int*)  d_in[6];
    const float* W1   = (const float*)d_in[7];
    const float* b1   = (const float*)d_in[8];
    const float* W2   = (const float*)d_in[9];
    const float* b2   = (const float*)d_in[10];
    float* out = (float*)d_out;

    char* ws = (char*)d_ws;
    float*          h1  = (float*)         (ws + OFF_H1);
    float*          h2  = (float*)         (ws + OFF_H2);
    unsigned short* A   = (unsigned short*)(ws + OFF_A);
    float*          HC  = (float*)         (ws + OFF_HC);
    unsigned short* W1T = (unsigned short*)(ws + OFF_W1T);
    unsigned short* W2T = (unsigned short*)(ws + OFF_W2T);

    const int B = 256;
    const long long nh = (long long)NNODES * FEAT;

    // ---- P1: h1 = propagate(x) ----
    k_zero_f32<<<(int)((nh + B - 1) / B), B, 0, stream>>>(h1, nh);
    k_propagate<<<(int)(((long long)NEDGES * 32 + B - 1) / B), B, 0, stream>>>(
        x, ei, ei + NEDGES, ew, h1);

    // ---- cluster1 gather + bf16 cast; W1 transpose-cast ----
    k_gather_cast_bf16<<<((NCLUST * 16) + B - 1) / B, B, 0, stream>>>(A, h1, rep1);
    k_transpose_cast<<<((FEAT * FEAT) + B - 1) / B, B, 0, stream>>>(W1T, W1, FEAT, FEAT);

    // ---- GEMM1 (bf16 WMMA, f32 acc) + bias + relu: HC[C,128] ----
    {
        int tiles = (NCLUST / 16) * (FEAT / 16);            // 30000 waves
        k_gemm_wmma<FEAT, true><<<(tiles * 32 + B - 1) / B, B, 0, stream>>>(
            A, W1T, b1, HC, NCLUST);
    }

    // ---- reconstruct into h1 (reused): h1[n] = HC[index1[n]] ----
    k_gather_f32_128<<<(int)(((long long)NNODES * 32 + B - 1) / B), B, 0, stream>>>(
        h1, HC, idx1);

    // ---- P2: h2 = propagate(h1) ----
    k_zero_f32<<<(int)((nh + B - 1) / B), B, 0, stream>>>(h2, nh);
    k_propagate<<<(int)(((long long)NEDGES * 32 + B - 1) / B), B, 0, stream>>>(
        h1, ei, ei + NEDGES, ew, h2);

    // ---- cluster2 gather + cast; W2 transpose-cast ----
    k_gather_cast_bf16<<<((NCLUST * 16) + B - 1) / B, B, 0, stream>>>(A, h2, rep2);
    k_transpose_cast<<<((FEAT * OUTF) + B - 1) / B, B, 0, stream>>>(W2T, W2, OUTF, FEAT);

    // ---- GEMM2 (bf16 WMMA, f32 acc) + bias: HC[C,64] ----
    {
        int tiles = (NCLUST / 16) * (OUTF / 16);            // 15000 waves
        k_gemm_wmma<OUTF, false><<<(tiles * 32 + B - 1) / B, B, 0, stream>>>(
            A, W2T, b2, HC, NCLUST);
    }

    // ---- log_softmax on compressed rows, then gather to out ----
    k_logsoftmax64<<<(int)(((long long)NCLUST * 32 + B - 1) / B), B, 0, stream>>>(
        HC, NCLUST);
    k_gather_f32_64<<<(int)(((long long)NNODES * 16 + B - 1) / B), B, 0, stream>>>(
        out, HC, idx2);
}
